// SpatioTemporalAttention_57466662420777
// MI455X (gfx1250) — matrix operand order repaired and checked by reading, not compiled, not run on hardware
//
#include <hip/hip_runtime.h>
#include <hip/hip_bf16.h>
#include <math.h>
#include <stdint.h>

typedef __attribute__((ext_vector_type(16))) _Float16 v16h;
typedef __attribute__((ext_vector_type(8)))  float    v8f;

// ---------------------------------------------------------------------------
// Async global->LDS (CDNA5, ASYNCcnt-tracked). GVS mode: mem = saddr + voff.
// LDS destination address = low 32 bits of generic shared pointer.
// ---------------------------------------------------------------------------
__device__ inline void async_b128(uint32_t lds_byte, uint32_t goff_bytes,
                                  const void* sbase) {
  asm volatile("global_load_async_to_lds_b128 %0, %1, %2 offset:0"
               :: "v"(lds_byte), "v"(goff_bytes),
                  "s"((uint64_t)(uintptr_t)sbase)
               : "memory");
}
__device__ inline uint32_t lds_of(const void* p) {
  return (uint32_t)(uintptr_t)p;
}

// ---------------------------------------------------------------------------
// WMMA fragment gathers (CDNA5 wave32 layouts, ISA 7.12.2), f32 LDS sources
// (convert-on-gather) and f16 LDS sources (for the P scratch).
// ---------------------------------------------------------------------------

// A fragment: 16x32 (MxK), source row-major [row][k], f32 in LDS.
__device__ inline v16h frag_a_16x32_f32(const float* base, int stride) {
  const int l = threadIdx.x & 31;
  const int row = l & 15;
  const int kh  = (l >> 4) * 8;
  const float* p = base + row * stride;
  v16h a;
#pragma unroll
  for (int i = 0; i < 8; ++i) a[i] = (_Float16)p[kh + i];
#pragma unroll
  for (int i = 0; i < 8; ++i) a[8 + i] = (_Float16)p[16 + kh + i];
  return a;
}

// A fragment from f16 source (P scratch).
__device__ inline v16h frag_a_16x32_f16(const _Float16* base, int stride) {
  const int l = threadIdx.x & 31;
  const int row = l & 15;
  const int kh  = (l >> 4) * 8;
  const _Float16* p = base + row * stride;
  v16h a;
#pragma unroll
  for (int i = 0; i < 8; ++i) a[i] = p[kh + i];
#pragma unroll
  for (int i = 0; i < 8; ++i) a[8 + i] = p[16 + kh + i];
  return a;
}

// B fragment: 32x16 (KxN) where source is stored [n][k] (B = src^T), f32.
__device__ inline v16h frag_b_nk_f32(const float* base, int stride) {
  const int l = threadIdx.x & 31;
  const int col = l & 15;
  const int k0  = (l >> 4) * 8;
  const float* p = base + col * stride + k0;
  v16h b;
#pragma unroll
  for (int i = 0; i < 16; ++i) b[i] = (_Float16)p[(i & 7) + ((i >> 3) << 4)];
  return b;
}

// B fragment: 32x16 (KxN) where source is stored [k][n] (K-major), f32.
__device__ inline v16h frag_b_kn_f32(const float* base, int stride) {
  const int l = threadIdx.x & 31;
  const int col = l & 15;
  const int k0  = (l >> 4) * 8;
  v16h b;
#pragma unroll
  for (int i = 0; i < 16; ++i) b[i] = (_Float16)base[(k0 + (i & 7) + ((i >> 3) << 4)) * stride + col];
  return b;
}

__device__ inline v8f wmma_f16(v16h a, v16h b, v8f c) {
  return __builtin_amdgcn_wmma_f32_16x16x32_f16(false, a, false, b, (short)0, c,
                                                false, false);
}

// ---------------------------------------------------------------------------
// GEMM:  C[M,N] = A[M,K] * W[N,K]^T + bias[N]   (K = 256)
// 256 threads = 8 waves; block tile 128x128; wave tile 32x64.
// Double-buffered async global->LDS pipeline (8 b128 transfers/thread/tile).
// ---------------------------------------------------------------------------
#define GTK 32

__global__ __launch_bounds__(256)
void gemm_bias_kernel(const float* __restrict__ A, const float* __restrict__ W,
                      const float* __restrict__ bias, float* __restrict__ C,
                      int M, int N, int K) {
  __shared__ float sA[2][128 * GTK];
  __shared__ float sW[2][128 * GTK];

  const int m0 = blockIdx.x * 128;
  const int n0 = blockIdx.y * 128;
  const int wave = threadIdx.x >> 5;
  const int wm = wave & 3;   // 4 waves along M
  const int wn = wave >> 2;  // 2 waves along N

  auto issue_tile = [&](int k0, int buf) {
#pragma unroll
    for (int t = 0; t < 4; ++t) {
      int id = threadIdx.x + t * 256;      // 0..1023
      int r  = id >> 3;                    // tile row
      int cg = (id & 7) * 4;               // 4-float group
      uint32_t ga = (uint32_t)((((size_t)(m0 + r)) * K + k0 + cg) * 4);
      async_b128(lds_of(&sA[buf][r * GTK + cg]), ga, A);
      uint32_t gw = (uint32_t)((((size_t)(n0 + r)) * K + k0 + cg) * 4);
      async_b128(lds_of(&sW[buf][r * GTK + cg]), gw, W);
    }
  };

  v8f acc[2][4];
#pragma unroll
  for (int i = 0; i < 2; ++i)
#pragma unroll
    for (int j = 0; j < 4; ++j) acc[i][j] = (v8f){0.f,0.f,0.f,0.f,0.f,0.f,0.f,0.f};

  issue_tile(0, 0);

  for (int k0 = 0; k0 < K; k0 += GTK) {
    const int cur = (k0 >> 5) & 1;
    const bool has_next = (k0 + GTK) < K;
    if (has_next) {
      issue_tile(k0 + GTK, cur ^ 1);
      asm volatile("s_wait_asynccnt 0x8" ::: "memory");  // retire tile k0 only
    } else {
      asm volatile("s_wait_asynccnt 0x0" ::: "memory");
    }
    __syncthreads();

    v16h af[2], bf[4];
#pragma unroll
    for (int i = 0; i < 2; ++i)
      af[i] = frag_a_16x32_f32(&sA[cur][(wm * 32 + i * 16) * GTK], GTK);
#pragma unroll
    for (int j = 0; j < 4; ++j)
      bf[j] = frag_b_nk_f32(&sW[cur][(wn * 64 + j * 16) * GTK], GTK);

#pragma unroll
    for (int i = 0; i < 2; ++i)
#pragma unroll
      for (int j = 0; j < 4; ++j) acc[i][j] = wmma_f16(af[i], bf[j], acc[i][j]);
    __syncthreads();  // protect buffer reuse against next prefetch
  }

  const int l = threadIdx.x & 31;
  const int colLane = l & 15;
  const int rowBase = (l >> 4) * 8;
#pragma unroll
  for (int i = 0; i < 2; ++i)
#pragma unroll
    for (int j = 0; j < 4; ++j) {
      int col = n0 + wn * 64 + j * 16 + colLane;
      float bv = bias[col];
#pragma unroll
      for (int e = 0; e < 8; ++e) {
        int row = m0 + wm * 32 + i * 16 + rowBase + e;
        C[(size_t)row * N + col] = acc[i][j][e] + bv;
      }
    }
}

// ---------------------------------------------------------------------------
// Spatial attention (flash): S=1024, dh=32, per (bt,h) x 64-query slab.
// qkv: [BT*S, 768], out: [BT*S, 256]. 128 threads = 4 waves x 16 query rows.
// K/V tiles (32x32 f32) double-buffered via async global->LDS.
// ---------------------------------------------------------------------------
#define SLD 40  // f16 P-scratch row stride

__global__ __launch_bounds__(128)
void spatial_attn_kernel(const float* __restrict__ qkv, float* __restrict__ out) {
  const int S = 1024, D3 = 768, Dm = 256, DH = 32;
  const int bh = blockIdx.x;           // bt*8 + h
  const int bt = bh >> 3;
  const int h  = bh & 7;
  const int q0 = blockIdx.y * 64;
  const int wave = threadIdx.x >> 5;
  const int l = threadIdx.x & 31;
  const size_t rowbase = (size_t)bt * S;
  const float scale = 0.1767766952966369f;  // 1/sqrt(32)

  __shared__ float    sKV[2][2][32 * 32];    // [buf][K|V][row*32+col] f32
  __shared__ _Float16 sP[4][16 * SLD];

  auto issue_kv = [&](int kb, int buf) {
#pragma unroll
    for (int t = 0; t < 4; ++t) {
      int id  = threadIdx.x + t * 128;   // 0..511
      int mat = id >> 8;                 // 0 = K, 1 = V
      int r   = (id >> 3) & 31;
      int cg  = (id & 7) * 4;
      uint32_t g = (uint32_t)(((rowbase + kb + r) * D3 +
                               Dm * (1 + mat) + h * DH + cg) * 4);
      async_b128(lds_of(&sKV[buf][mat][r * 32 + cg]), g, qkv);
    }
  };

  // Q fragment (scale folded in), A-layout 16x32, from global f32
  v16h qf;
  {
    int row = l & 15, kh = (l >> 4) * 8;
    const float* qp = qkv + (rowbase + q0 + wave * 16 + row) * D3 + h * DH;
#pragma unroll
    for (int i = 0; i < 8; ++i) qf[i] = (_Float16)(qp[kh + i] * scale);
#pragma unroll
    for (int i = 0; i < 8; ++i) qf[8 + i] = (_Float16)(qp[16 + kh + i] * scale);
  }

  float m[8], lsum[8];
  v8f o0 = (v8f){0.f,0.f,0.f,0.f,0.f,0.f,0.f,0.f}, o1 = o0;
#pragma unroll
  for (int e = 0; e < 8; ++e) { m[e] = -1e30f; lsum[e] = 0.f; }

  const int rowB = (l >> 4) * 8;
  const int colL = l & 15;

  issue_kv(0, 0);

  for (int kb = 0; kb < S; kb += 32) {
    const int cur = (kb >> 5) & 1;
    const bool has_next = (kb + 32) < S;
    if (has_next) {
      issue_kv(kb + 32, cur ^ 1);
      asm volatile("s_wait_asynccnt 0x4" ::: "memory");
    } else {
      asm volatile("s_wait_asynccnt 0x0" ::: "memory");
    }
    __syncthreads();

    v8f z = (v8f){0.f,0.f,0.f,0.f,0.f,0.f,0.f,0.f};
    v16h b0 = frag_b_nk_f32(&sKV[cur][0][0], 32);        // keys kb..kb+15
    v16h b1 = frag_b_nk_f32(&sKV[cur][0][16 * 32], 32);  // keys kb+16..+31
    v8f s0 = wmma_f16(qf, b0, z);
    v8f s1 = wmma_f16(qf, b1, z);

    // online softmax per row (row stats redundant across the 16-lane half)
#pragma unroll
    for (int e = 0; e < 8; ++e) {
      float rmax = fmaxf(s0[e], s1[e]);
#pragma unroll
      for (int off = 1; off < 16; off <<= 1)
        rmax = fmaxf(rmax, __shfl_xor(rmax, off, 32));
      float mnew = fmaxf(m[e], rmax);
      float alpha = __expf(m[e] - mnew);
      float p0 = __expf(s0[e] - mnew);
      float p1 = __expf(s1[e] - mnew);
      float psum = p0 + p1;
#pragma unroll
      for (int off = 1; off < 16; off <<= 1)
        psum += __shfl_xor(psum, off, 32);
      lsum[e] = lsum[e] * alpha + psum;
      m[e] = mnew;
      o0[e] *= alpha;
      o1[e] *= alpha;
      sP[wave][(rowB + e) * SLD + colL]      = (_Float16)p0;
      sP[wave][(rowB + e) * SLD + 16 + colL] = (_Float16)p1;
    }

    v16h pf = frag_a_16x32_f16(&sP[wave][0], SLD);     // P: 16 x 32(keys)
    v16h v0 = frag_b_kn_f32(&sKV[cur][1][0], 32);      // V: keys x dh[0..15]
    v16h v1 = frag_b_kn_f32(&sKV[cur][1][16], 32);     // V: keys x dh[16..31]
    o0 = wmma_f16(pf, v0, o0);
    o1 = wmma_f16(pf, v1, o1);
    __syncthreads();
  }

#pragma unroll
  for (int e = 0; e < 8; ++e) {
    float inv = 1.0f / lsum[e];
    size_t row = rowbase + q0 + wave * 16 + rowB + e;
    out[row * Dm + h * DH + colL]      = o0[e] * inv;
    out[row * Dm + h * DH + 16 + colL] = o1[e] * inv;
  }
}

// ---------------------------------------------------------------------------
// Temporal attention: S=16, dh=32, one wave per (b,n,h). 16384 waves total.
// qkv: [B*T*N, 768] in [B,T,N,*] order; out: [B*T*N, 256]
// ---------------------------------------------------------------------------
__global__ __launch_bounds__(256)
void temporal_attn_kernel(const float* __restrict__ qkv, float* __restrict__ out) {
  const int T = 16, Nn = 1024, D3 = 768, Dm = 256, DH = 32;
  const int wave = threadIdx.x >> 5;
  const int l = threadIdx.x & 31;
  const int wid = blockIdx.x * 8 + wave;  // 0..16383
  const int h = wid & 7;
  const int n = (wid >> 3) & 1023;
  const int b = wid >> 13;
  const float scale = 0.1767766952966369f;

  __shared__ _Float16 sP[8][16 * 20];

  auto rowOf = [&](int t) -> size_t { return (size_t)((b * T + t) * Nn + n); };

  v16h qf;
  {
    int row = l & 15, kh = (l >> 4) * 8;
    const float* qp = qkv + rowOf(row) * D3 + h * DH;
#pragma unroll
    for (int i = 0; i < 8; ++i) qf[i] = (_Float16)(qp[kh + i] * scale);
#pragma unroll
    for (int i = 0; i < 8; ++i) qf[8 + i] = (_Float16)(qp[16 + kh + i] * scale);
  }
  v16h kf;
  {
    int col = l & 15, k0 = (l >> 4) * 8;
    const float* kp = qkv + rowOf(col) * D3 + Dm + h * DH + k0;
#pragma unroll
    for (int i = 0; i < 16; ++i) kf[i] = (_Float16)kp[(i & 7) + ((i >> 3) << 4)];
  }

  v8f z = (v8f){0.f,0.f,0.f,0.f,0.f,0.f,0.f,0.f};
  v8f s = wmma_f16(qf, kf, z);

  const int rowB = (l >> 4) * 8;
  const int colL = l & 15;
  float lsum[8];
#pragma unroll
  for (int e = 0; e < 8; ++e) {
    float rmax = s[e];
#pragma unroll
    for (int off = 1; off < 16; off <<= 1)
      rmax = fmaxf(rmax, __shfl_xor(rmax, off, 32));
    float p = __expf(s[e] - rmax);
    float psum = p;
#pragma unroll
    for (int off = 1; off < 16; off <<= 1)
      psum += __shfl_xor(psum, off, 32);
    lsum[e] = psum;
    sP[wave][(rowB + e) * 20 + colL] = (_Float16)p;
  }

  v16h pf;  // P as A fragment 16x32, K 16..31 zero-padded
  {
    int row = l & 15, kh = (l >> 4) * 8;
#pragma unroll
    for (int i = 0; i < 8; ++i) {
      int k = kh + i;
      pf[i] = (k < 16) ? sP[wave][row * 20 + k] : (_Float16)0.f;
    }
#pragma unroll
    for (int i = 0; i < 8; ++i) {
      int k = 16 + kh + i;
      pf[8 + i] = (k < 16) ? sP[wave][row * 20 + k] : (_Float16)0.f;
    }
  }
  v16h vf0, vf1;  // V: B[k=t][col=d], t>=16 zero
  {
    int col = l & 15, k0 = (l >> 4) * 8;
#pragma unroll
    for (int i = 0; i < 16; ++i) {
      int k = k0 + (i & 7) + ((i >> 3) << 4);
      if (k < 16) {
        const float* vp = qkv + rowOf(k) * D3 + 2 * Dm + h * DH;
        vf0[i] = (_Float16)vp[col];
        vf1[i] = (_Float16)vp[16 + col];
      } else {
        vf0[i] = (_Float16)0.f;
        vf1[i] = (_Float16)0.f;
      }
    }
  }
  v8f o0 = wmma_f16(pf, vf0, z);
  v8f o1 = wmma_f16(pf, vf1, z);

#pragma unroll
  for (int e = 0; e < 8; ++e) {
    float inv = 1.0f / lsum[e];
    size_t row = rowOf(rowB + e);
    out[row * Dm + h * DH + colL]      = o0[e] * inv;
    out[row * Dm + h * DH + 16 + colL] = o1[e] * inv;
  }
}

// ---------------------------------------------------------------------------
// Orchestration
// ---------------------------------------------------------------------------
extern "C" void kernel_launch(void* const* d_in, const int* in_sizes, int n_in,
                              void* d_out, int out_size, void* d_ws, size_t ws_size,
                              hipStream_t stream) {
  (void)in_sizes; (void)n_in; (void)out_size; (void)ws_size;
  const float* x      = (const float*)d_in[0];
  const float* s_wqkv = (const float*)d_in[1];
  const float* s_bqkv = (const float*)d_in[2];
  const float* s_wo   = (const float*)d_in[3];
  const float* s_bo   = (const float*)d_in[4];
  const float* t_wqkv = (const float*)d_in[5];
  const float* t_bqkv = (const float*)d_in[6];
  const float* t_wo   = (const float*)d_in[7];
  const float* t_bo   = (const float*)d_in[8];
  const float* p_w    = (const float*)d_in[9];
  const float* p_b    = (const float*)d_in[10];

  const int M = 2 * 16 * 1024;  // 32768 tokens
  float* qkv  = (float*)d_ws;                       // M x 768
  float* attn = qkv + (size_t)M * 768;              // M x 256
  float* proj = attn + (size_t)M * 256;             // M x 256
  float* outp = (float*)d_out;                      // M x 256

  dim3 g3(M / 128, 768 / 128), g1(M / 128, 256 / 128);

  gemm_bias_kernel<<<g3, 256, 0, stream>>>(x, s_wqkv, s_bqkv, qkv, M, 768, 256);
  spatial_attn_kernel<<<dim3(256, 16), 128, 0, stream>>>(qkv, attn);
  gemm_bias_kernel<<<g1, 256, 0, stream>>>(attn, s_wo, s_bo, proj, M, 256, 256);
  gemm_bias_kernel<<<g3, 256, 0, stream>>>(proj, t_wqkv, t_bqkv, qkv, M, 768, 256);
  temporal_attn_kernel<<<2048, 256, 0, stream>>>(qkv, attn);
  gemm_bias_kernel<<<g1, 256, 0, stream>>>(attn, t_wo, t_bo, proj, M, 256, 256);
  gemm_bias_kernel<<<g1, 256, 0, stream>>>(proj, p_w, p_b, outp, M, 256, 256);
}
